// ShapeRetrieval_12171937316901
// MI455X (gfx1250) — matrix operand order repaired and checked by reading, not compiled, not run on hardware
//
#include <hip/hip_runtime.h>
#include <math.h>
#include <limits.h>

typedef __attribute__((ext_vector_type(16))) __bf16 v16bf;
typedef __attribute__((ext_vector_type(8)))  __bf16 v8bf;
typedef __attribute__((ext_vector_type(8)))  float  v8f;
typedef unsigned int __attribute__((ext_vector_type(4))) v4u;
typedef int          __attribute__((ext_vector_type(4))) v4i;
typedef int          __attribute__((ext_vector_type(8))) v8i;

#define D_DIM 256
#define KSTEPS 8                    // 256 / 32
#define TILE_ELEMS (16 * D_DIM)     // bf16 elements per 16-row gallery tile (8 KB)
#define STILE 2                     // gallery tiles per TDM supertile
#define SUPER_PER_SPLIT 25          // supertiles per gallery split (=50 tiles)

// ---------------------------------------------------------------------------
// Pass 1a: L2-normalize gallery rows, emit bf16 in WMMA *B-fragment* order.
// B (32x16 bf16): lane = ((k>>4)&1)*16 + n, elem-in-lane = k&15.
// ---------------------------------------------------------------------------
__global__ void norm_gallery_kernel(const float* __restrict__ g,
                                    __bf16* __restrict__ gfrag, int M) {
    const int wave = threadIdx.x >> 5;
    const int lane = threadIdx.x & 31;
    const int row  = blockIdx.x * 8 + wave;
    if (row >= M) return;

    const float* r = g + (size_t)row * D_DIM + lane * 8;
    float4 v0 = *(const float4*)r;
    float4 v1 = *(const float4*)(r + 4);
    float ss = v0.x*v0.x + v0.y*v0.y + v0.z*v0.z + v0.w*v0.w
             + v1.x*v1.x + v1.y*v1.y + v1.z*v1.z + v1.w*v1.w;
    #pragma unroll
    for (int m = 1; m < 32; m <<= 1) ss += __shfl_xor(ss, m, 32);
    const float scale = 1.0f / fmaxf(sqrtf(ss), 1e-12f);

    v8bf o;
    o[0]=(__bf16)(v0.x*scale); o[1]=(__bf16)(v0.y*scale);
    o[2]=(__bf16)(v0.z*scale); o[3]=(__bf16)(v0.w*scale);
    o[4]=(__bf16)(v1.x*scale); o[5]=(__bf16)(v1.y*scale);
    o[6]=(__bf16)(v1.z*scale); o[7]=(__bf16)(v1.w*scale);

    const int mt = row >> 4;
    const int kt = lane >> 2;
    const int fl = (((lane >> 1) & 1) << 4) + (row & 15);
    const int e0 = (lane & 1) << 3;
    __bf16* dst = gfrag + ((((size_t)mt * KSTEPS + kt) * 32 + fl) * 16 + e0);
    *(v8bf*)dst = o;
}

// ---------------------------------------------------------------------------
// Pass 1b: L2-normalize query rows, emit bf16 in WMMA *A-fragment* order.
// A (16x32 bf16): lane = ((k>>4)&1)*16 + m, elem-in-lane = k&15.
// ---------------------------------------------------------------------------
__global__ void norm_query_kernel(const float* __restrict__ q,
                                  __bf16* __restrict__ qfrag, int N) {
    const int wave = threadIdx.x >> 5;
    const int lane = threadIdx.x & 31;
    const int row  = blockIdx.x * 8 + wave;
    if (row >= N) return;

    const float* r = q + (size_t)row * D_DIM + lane * 8;
    float4 v0 = *(const float4*)r;
    float4 v1 = *(const float4*)(r + 4);
    float ss = v0.x*v0.x + v0.y*v0.y + v0.z*v0.z + v0.w*v0.w
             + v1.x*v1.x + v1.y*v1.y + v1.z*v1.z + v1.w*v1.w;
    #pragma unroll
    for (int m = 1; m < 32; m <<= 1) ss += __shfl_xor(ss, m, 32);
    const float scale = 1.0f / fmaxf(sqrtf(ss), 1e-12f);

    v8bf o;
    o[0]=(__bf16)(v0.x*scale); o[1]=(__bf16)(v0.y*scale);
    o[2]=(__bf16)(v0.z*scale); o[3]=(__bf16)(v0.w*scale);
    o[4]=(__bf16)(v1.x*scale); o[5]=(__bf16)(v1.y*scale);
    o[6]=(__bf16)(v1.z*scale); o[7]=(__bf16)(v1.w*scale);

    const int qt = row >> 4;
    const int kt = lane >> 2;
    const int fl = (((lane >> 1) & 1) << 4) + (row & 15);
    const int e0 = (lane & 1) << 3;
    __bf16* dst = qfrag + ((((size_t)qt * KSTEPS + kt) * 32 + fl) * 16 + e0);
    *(v8bf*)dst = o;
}

// ---------------------------------------------------------------------------
// TDM: stage `elems` bf16 (up to one 16KB supertile, already in B-fragment
// order) into LDS. D# per ISA §8: group0 {count=1, lds_addr, global_addr,
// type=2}; group1 {data_size=2B, tensor_dim0=tile_dim0=elems, tile_dim1=1}.
// ---------------------------------------------------------------------------
__device__ __forceinline__ void tdm_load(const __bf16* gsrc, __bf16* ldsDst,
                                         int elems) {
    const unsigned long long ga = (unsigned long long)(size_t)gsrc;
    const unsigned int lds_off  = (unsigned int)(size_t)ldsDst; // addr[31:0]

    v4u g0;
    g0[0] = 1u;                                           // count=1 (valid, user)
    g0[1] = lds_off;                                      // lds_addr
    g0[2] = (unsigned int)(ga & 0xFFFFFFFFu);             // global_addr[31:0]
    g0[3] = (unsigned int)((ga >> 32) & 0x01FFFFFFu)      // global_addr[56:32]
          | (2u << 30);                                   // type=2 ("image")

    v8i g1;
    g1[0] = (1 << 16);                                    // data_size=1 -> 2 bytes
    g1[1] = (int)((elems & 0xFFFF) << 16);                // tensor_dim0[15:0]
    g1[2] = (int)(((elems >> 16) & 0xFFFF) | (1 << 16));  // dim0[31:16], dim1=1
    g1[3] = (int)((elems & 0xFFFF) << 16);                // tile_dim0 = elems
    g1[4] = 1;                                            // tile_dim1=1, tile_dim2=0
    g1[5] = elems;                                        // tensor_dim0_stride
    g1[6] = 0;
    g1[7] = 0;

    v4i gz = {0, 0, 0, 0};
#if defined(__clang_major__) && __clang_major__ >= 23
    v8i gz8 = {0, 0, 0, 0, 0, 0, 0, 0};
    __builtin_amdgcn_tensor_load_to_lds(g0, g1, gz, gz, gz8, 0);
#else
    __builtin_amdgcn_tensor_load_to_lds(g0, g1, gz, gz, 0);
#endif
}

// ---------------------------------------------------------------------------
// Pass 2: fused bf16 WMMA GEMM + running max/argmax.
// TDM double-buffers 2-tile supertiles in LDS (one barrier + one TDM per
// supertile); each wave keeps its A fragments resident and runs two
// independent 8-WMMA accumulation chains, preloading all B fragments into
// registers so LDS latency is paid once, not per k-step.
// ---------------------------------------------------------------------------
__global__ void sim_argmax_kernel(const __bf16* __restrict__ qfrag,
                                  const __bf16* __restrict__ gfrag,
                                  float* __restrict__ pmax,
                                  int*   __restrict__ pidx,
                                  int N, int M, int gtiles, int S) {
    __shared__ __attribute__((aligned(32))) __bf16 bstage[2][STILE * TILE_ELEMS];

    const int lane   = threadIdx.x & 31;
    const int wave   = threadIdx.x >> 5;
    const int split  = blockIdx.x;
    const int qtiles = N >> 4;
    int qt = blockIdx.y * 8 + wave;
    const bool qvalid = qt < qtiles;
    if (!qvalid) qt = qtiles - 1;        // clamp; store predicated at end

    // Resident A fragments for all 8 k-steps.
    const v16bf* qf = (const v16bf*)qfrag;
    v16bf a[KSTEPS];
    #pragma unroll
    for (int k = 0; k < KSTEPS; ++k)
        a[k] = qf[((size_t)qt * KSTEPS + k) * 32 + lane];

    float runmax[8];
    int   runidx[8];
    #pragma unroll
    for (int j = 0; j < 8; ++j) { runmax[j] = -INFINITY; runidx[j] = 0; }

    const int stiles = (gtiles + STILE - 1) / STILE;
    const int st0 = split * SUPER_PER_SPLIT;
    int st1 = st0 + SUPER_PER_SPLIT; if (st1 > stiles) st1 = stiles;

    const int total_elems = gtiles * TILE_ELEMS;
    auto super_elems = [&](int st) {
        int e = total_elems - st * (STILE * TILE_ELEMS);
        return e > STILE * TILE_ELEMS ? STILE * TILE_ELEMS : e;
    };

    // Prologue: async-stage supertile st0 into buffer 0 (wave 0 drives TDM).
    if (wave == 0 && st0 < st1)
        tdm_load(gfrag + (size_t)st0 * (STILE * TILE_ELEMS), &bstage[0][0],
                 super_elems(st0));

    for (int st = st0; st < st1; ++st) {
        const int cur = st & 1;
        if (wave == 0)
            __builtin_amdgcn_s_wait_tensorcnt(0);   // TDM(st) landed in LDS
        __syncthreads();                            // broadcast; closes WAR on
                                                    // bstage[1-cur] readers
        if (wave == 0 && st + 1 < st1)              // overlap TDM(st+1)
            tdm_load(gfrag + (size_t)(st + 1) * (STILE * TILE_ELEMS),
                     &bstage[1 - cur][0], super_elems(st + 1));

        // Preload ALL B fragments for both tiles, then run two independent
        // accumulation chains (LDS latency paid once; chains interleave).
        const v16bf* bp = (const v16bf*)&bstage[cur][0] + lane;
        v16bf b0[KSTEPS], b1[KSTEPS];
        #pragma unroll
        for (int k = 0; k < KSTEPS; ++k) b0[k] = bp[k * 32];
        #pragma unroll
        for (int k = 0; k < KSTEPS; ++k) b1[k] = bp[KSTEPS * 32 + k * 32];

        v8f c0 = {}, c1 = {};
        #pragma unroll
        for (int k = 0; k < KSTEPS; ++k) {
            c0 = __builtin_amdgcn_wmma_f32_16x16x32_bf16(
                     false, a[k], false, b0[k], (short)0, c0, false, false);
            c1 = __builtin_amdgcn_wmma_f32_16x16x32_bf16(
                     false, a[k], false, b1[k], (short)0, c1, false, false);
        }

        const int gcol0 = (st * STILE) * 16 + (lane & 15);
        const int gcol1 = gcol0 + 16;
        #pragma unroll
        for (int j = 0; j < 8; ++j) {
            if (c0[j] > runmax[j]) { runmax[j] = c0[j]; runidx[j] = gcol0; }
            if (gcol1 < M && c1[j] > runmax[j]) {
                runmax[j] = c1[j]; runidx[j] = gcol1;
            }
        }
    }

    // Reduce across the 16 gallery columns held by each half-wave.
    #pragma unroll
    for (int j = 0; j < 8; ++j) {
        float m = runmax[j];
        int   i = runidx[j];
        #pragma unroll
        for (int off = 1; off < 16; off <<= 1) {    // stays inside each half
            float om = __shfl_xor(m, off, 32);
            int   oi = __shfl_xor(i, off, 32);
            if (om > m || (om == m && oi < i)) { m = om; i = oi; }
        }
        if (qvalid && (lane & 15) == 0) {
            const int mrow = ((lane >> 4) << 3) + j;
            const int q    = qt * 16 + mrow;
            pmax[(size_t)q * S + split] = m;
            pidx[(size_t)q * S + split] = i;
        }
    }
}

// ---------------------------------------------------------------------------
// Pass 3: reduce over gallery splits, gather shape ids, emit outputs as
// (idx[N], max_sim[N], retrieved[N]) flattened to float.
// ---------------------------------------------------------------------------
__global__ void finalize_kernel(const float* __restrict__ pmax,
                                const int*   __restrict__ pidx,
                                const int*   __restrict__ table,
                                float* __restrict__ out, int N, int S) {
    const int q = blockIdx.x * blockDim.x + threadIdx.x;
    if (q >= N) return;
    float m = -INFINITY;
    int   idx = INT_MAX;
    for (int s = 0; s < S; ++s) {
        const float om = pmax[(size_t)q * S + s];
        const int   oi = pidx[(size_t)q * S + s];
        if (om > m || (om == m && oi < idx)) { m = om; idx = oi; }
    }
    out[q]         = (float)idx;
    out[N + q]     = m;
    out[2 * N + q] = (float)table[idx];
}

// ---------------------------------------------------------------------------
extern "C" void kernel_launch(void* const* d_in, const int* in_sizes, int n_in,
                              void* d_out, int out_size, void* d_ws, size_t ws_size,
                              hipStream_t stream) {
    const float* q_in  = (const float*)d_in[0];
    const float* g_in  = (const float*)d_in[1];
    const int*   table = (const int*)d_in[2];

    const int N = in_sizes[0] / D_DIM;   // 2048
    const int M = in_sizes[1] / D_DIM;   // 100000
    const int gtiles = M / 16;           // 6250
    const int stiles = (gtiles + STILE - 1) / STILE;               // 3125
    const int S = (stiles + SUPER_PER_SPLIT - 1) / SUPER_PER_SPLIT; // 125
    const int qgroups = (N + 127) / 128; // 16

    // Workspace partition (256B aligned regions).
    char* ws = (char*)d_ws;
    size_t off = 0;
    __bf16* gfrag = (__bf16*)(ws + off);
    off += ((size_t)M * D_DIM * sizeof(__bf16) + 255) & ~(size_t)255;
    __bf16* qfrag = (__bf16*)(ws + off);
    off += ((size_t)N * D_DIM * sizeof(__bf16) + 255) & ~(size_t)255;
    float* pmax = (float*)(ws + off);
    off += ((size_t)N * S * sizeof(float) + 255) & ~(size_t)255;
    int* pidx = (int*)(ws + off);

    norm_gallery_kernel<<<(M + 7) / 8, 256, 0, stream>>>(g_in, gfrag, M);
    norm_query_kernel<<<(N + 7) / 8, 256, 0, stream>>>(q_in, qfrag, N);

    dim3 grid(S, qgroups);
    sim_argmax_kernel<<<grid, 256, 0, stream>>>(qfrag, gfrag, pmax, pidx,
                                                N, M, gtiles, S);

    finalize_kernel<<<(N + 255) / 256, 256, 0, stream>>>(pmax, pidx, table,
                                                         (float*)d_out, N, S);
}
